// ODELSTM_75531294868004
// MI455X (gfx1250) — matrix operand order repaired and checked by reading, not compiled, not run on hardware
//
#include <hip/hip_runtime.h>

// ---------------------------------------------------------------------------
// ODE-LSTM step pipeline for MI455X (gfx1250, wave32, WMMA bf16).
// GEMMs use v_wmma_f32_16x16x32_bf16 with a 4x2-tile (64x32) region per wave:
// per K-chunk, 2 B-fragments + 4 A-fragments (6 loads) feed 8 WMMAs
// (1.5 b128 loads per WMMA), keeping the XDL pipe fed instead of load-bound.
// ---------------------------------------------------------------------------

typedef __bf16 bf16_t;
typedef __attribute__((ext_vector_type(16))) __bf16 v16bf;
typedef __attribute__((ext_vector_type(8)))  float  v8f;

#define NB_   4096        // B
#define NT_   100         // T
#define NP_   2048
#define ND_   32          // D
#define NS_   4           // S
#define NH_   256         // H
#define DT_   0.1f
#define EPS_  1e-8f
#define NDS_  (ND_*NS_)   // 128 = D*S
#define KCAT_ (NDS_+NH_)  // 384
#define G4H_  (4*NH_)     // 1024
#define MT_   4           // row sub-tiles per wave (64 rows)
#define ROWS_PER_BLOCK_ (8 * MT_ * 16)   // 8 waves * 64 rows = 512

__device__ __forceinline__ bf16_t to_bf16(float x) { return (bf16_t)x; }
__device__ __forceinline__ float  sigm(float x)    { return 1.0f / (1.0f + __expf(-x)); }

// --- A fragment: 16x32 bf16 tile, row-major source, contiguous k order (k = 16h + s), same as B.
// lane 0-15  : row=lane,    elems 0..15 -> K=k0+0..15
// lane 16-31 : row=lane-16, elems 0..15 -> K=k0+16..31
__device__ __forceinline__ v16bf load_fragA(const bf16_t* base, int ld, int row0, int k0, int lane) {
    int r  = lane & 15;
    int kh = lane >> 4;
    const bf16_t* p = base + (size_t)(row0 + r) * ld + k0 + kh * 16;
    v16bf f;
#pragma unroll
    for (int i = 0; i < 16; ++i) f[i] = p[i];
    return f;
}

// --- B fragment: 32x16 bf16 (K x N). Column n of B == row n of row-major W(N x K).
// lane 0-15 : col=lane,    elems 0..15 -> K=k0+0..15
// lane 16-31: col=lane-16, elems 0..15 -> K=k0+16..31
__device__ __forceinline__ v16bf load_fragB(const bf16_t* W, int ldk, int n0, int k0, int lane) {
    int n  = lane & 15;
    int kh = lane >> 4;
    const bf16_t* p = W + (size_t)(n0 + n) * ldk + k0 + kh * 16;
    v16bf f;
#pragma unroll
    for (int i = 0; i < 16; ++i) f[i] = p[i];
    return f;
}

__device__ __forceinline__ v8f wmma_bf16(v16bf a, v16bf b, v8f c) {
    return __builtin_amdgcn_wmma_f32_16x16x32_bf16(false, a, false, b, (short)0, c, false, false);
}

// 4x2 tile GEMM: C(64x32 at row0,col0) = A(64xK) * W^T(Kx32) + bias.
// b0/b1 = per-lane bias for column sub-tiles 0/1.
// C/D 16x16 f32 layout: lane 0-15: (M=r, N=lane); lane 16-31: (M=8+r, N=lane-16).
__device__ __forceinline__ void gemm_4x2(const bf16_t* A, int lda, const bf16_t* W, int ldk,
                                         int row0, int col0, int K, float b0, float b1,
                                         v8f acc[MT_][2], int lane) {
#pragma unroll
    for (int m = 0; m < MT_; ++m)
#pragma unroll
        for (int i = 0; i < 8; ++i) { acc[m][0][i] = b0; acc[m][1][i] = b1; }
    for (int k = 0; k < K; k += 32) {
        v16bf fb0 = load_fragB(W, ldk, col0,      k, lane);
        v16bf fb1 = load_fragB(W, ldk, col0 + 16, k, lane);
#pragma unroll
        for (int m = 0; m < MT_; ++m) {
            v16bf fa = load_fragA(A, lda, row0 + m * 16, k, lane);
            acc[m][0] = wmma_bf16(fa, fb0, acc[m][0]);
            acc[m][1] = wmma_bf16(fa, fb1, acc[m][1]);
        }
    }
}

// ---------------------------------------------------------------------------
// Kernels
// ---------------------------------------------------------------------------

__global__ void k_init(float* h, float* c, bf16_t* hb, float* accum, int n) {
    int i = blockIdx.x * blockDim.x + threadIdx.x;
    if (i < n) { h[i] = 0.f; c[i] = 0.f; hb[i] = to_bf16(0.f); }
    if (i < 4) accum[i] = 0.f;
}

// Pack all weights to bf16 (once per call). Wcat = [W_ih | W_hh] (1024 x 384).
__global__ void k_pack(const float* Wo1, const float* Wo2, const float* Wp1, const float* Wp2,
                       const float* Wih, const float* Whh,
                       bf16_t* Wo1b, bf16_t* Wo2b, bf16_t* Wp1b, bf16_t* Wp2b, bf16_t* Wcatb) {
    int i = blockIdx.x * blockDim.x + threadIdx.x;
    if (i < NH_ * NH_) {
        Wo1b[i] = to_bf16(Wo1[i]);
        Wo2b[i] = to_bf16(Wo2[i]);
        Wp1b[i] = to_bf16(Wp1[i]);
    }
    if (i < 32 * NH_)  // pad (D-1)=31 rows to 32 with zeros
        Wp2b[i] = (i < 31 * NH_) ? to_bf16(Wp2[i]) : to_bf16(0.f);
    if (i < G4H_ * KCAT_) {
        int r = i / KCAT_, k = i % KCAT_;
        float v = (k < NDS_) ? Wih[r * NDS_ + k] : Whh[r * NH_ + (k - NDS_)];
        Wcatb[i] = to_bf16(v);
    }
}

// U = tanh(h @ Wo1^T + bo1)  (4096x256), bf16 out
__global__ void k_gemm_tanh(const bf16_t* A, const bf16_t* W, const float* bias, bf16_t* Ub) {
    int lane = threadIdx.x & 31, wave = threadIdx.x >> 5;
    int row0 = (blockIdx.x * 8 + wave) * (MT_ * 16);
    int col0 = blockIdx.y * 32;
    int cn = lane & 15, rb = (lane >> 4) * 8;
    float b0 = bias[col0 + cn], b1 = bias[col0 + 16 + cn];
    v8f acc[MT_][2];
    gemm_4x2(A, NH_, W, NH_, row0, col0, NH_, b0, b1, acc, lane);
#pragma unroll
    for (int m = 0; m < MT_; ++m)
#pragma unroll
        for (int r = 0; r < 8; ++r) {
            size_t ro = (size_t)(row0 + m * 16 + rb + r) * NH_ + col0 + cn;
            Ub[ro]      = to_bf16(tanhf(acc[m][0][r]));
            Ub[ro + 16] = to_bf16(tanhf(acc[m][1][r]));
        }
}

// h += DT * tanh(U @ Wo2^T + bo2); refresh f32 master + bf16 shadow
__global__ void k_gemm_ode(const bf16_t* Ub, const bf16_t* W, const float* bias,
                           float* h, bf16_t* hb) {
    int lane = threadIdx.x & 31, wave = threadIdx.x >> 5;
    int row0 = (blockIdx.x * 8 + wave) * (MT_ * 16);
    int col0 = blockIdx.y * 32;
    int cn = lane & 15, rb = (lane >> 4) * 8;
    float b0 = bias[col0 + cn], b1 = bias[col0 + 16 + cn];
    v8f acc[MT_][2];
    gemm_4x2(Ub, NH_, W, NH_, row0, col0, NH_, b0, b1, acc, lane);
#pragma unroll
    for (int m = 0; m < MT_; ++m)
#pragma unroll
        for (int r = 0; r < 8; ++r) {
            size_t o0 = (size_t)(row0 + m * 16 + rb + r) * NH_ + col0 + cn;
            size_t o1 = o0 + 16;
            float h0 = h[o0] + DT_ * tanhf(acc[m][0][r]);
            float h1 = h[o1] + DT_ * tanhf(acc[m][1][r]);
            h[o0] = h0; hb[o0] = to_bf16(h0);
            h[o1] = h1; hb[o1] = to_bf16(h1);
        }
}

// Acat (2048 x 384) = [ bf16(X[t])  |  hb[batch_idx[t]] ]
__global__ void k_gather(const float* X, const int* bidx, const bf16_t* hb, bf16_t* Acat, int t) {
    int i = blockIdx.x * blockDim.x + threadIdx.x;
    if (i >= NP_ * KCAT_) return;
    int p = i / KCAT_, k = i % KCAT_;
    bf16_t v;
    if (k < NDS_) {
        v = to_bf16(X[((size_t)t * NP_ + p) * NDS_ + k]);
    } else {
        int b = bidx[t * NP_ + p];
        v = hb[(size_t)b * NH_ + (k - NDS_)];
    }
    Acat[i] = v;
}

// Q = relu(h[idx] @ Wp1^T + bp1)  (2048x256) — A is Acat's hidden slice (lda=384)
__global__ void k_gemm_relu(const bf16_t* Acat, const bf16_t* W, const float* bias, bf16_t* Qb) {
    int lane = threadIdx.x & 31, wave = threadIdx.x >> 5;
    int row0 = (blockIdx.x * 8 + wave) * (MT_ * 16);
    int col0 = blockIdx.y * 32;
    int cn = lane & 15, rb = (lane >> 4) * 8;
    float b0 = bias[col0 + cn], b1 = bias[col0 + 16 + cn];
    v8f acc[MT_][2];
    gemm_4x2(Acat + NDS_, KCAT_, W, NH_, row0, col0, NH_, b0, b1, acc, lane);
#pragma unroll
    for (int m = 0; m < MT_; ++m)
#pragma unroll
        for (int r = 0; r < 8; ++r) {
            size_t ro = (size_t)(row0 + m * 16 + rb + r) * NH_ + col0 + cn;
            Qb[ro]      = to_bf16(fmaxf(acc[m][0][r], 0.f));
            Qb[ro + 16] = to_bf16(fmaxf(acc[m][1][r], 0.f));
        }
}

// P = Q @ Wp2^T + bp2 (2048 x 31, padded to 32); fused masked mse/mae/mape/tot.
// One wave covers 64 rows x all 32 output cols (col 31 is zero padding).
__global__ void k_gemm_err(const bf16_t* Qb, const bf16_t* W, const float* bias,
                           const float* X, const float* Mm, int t, float* accum) {
    int lane = threadIdx.x & 31, wave = threadIdx.x >> 5;
    int row0 = (blockIdx.x * 8 + wave) * (MT_ * 16);
    int cn = lane & 15, rb = (lane >> 4) * 8;
    float b0 = bias[cn];                              // cols 0..15 all valid
    float b1 = (16 + cn < 31) ? bias[16 + cn] : 0.f;  // col 31 padded
    v8f acc[MT_][2];
    gemm_4x2(Qb, NH_, W, NH_, row0, 0, NH_, b0, b1, acc, lane);

    float s0 = 0.f, s1 = 0.f, s2 = 0.f, s3 = 0.f;
#pragma unroll
    for (int ct = 0; ct < 2; ++ct) {
        int col = ct * 16 + cn;            // output feature j (valid: 0..30)
        if (col < 31) {
            int d = col + 1;               // D index 1..31
#pragma unroll
            for (int m = 0; m < MT_; ++m)
#pragma unroll
                for (int r = 0; r < 8; ++r) {
                    float pv = acc[m][ct][r];
                    size_t xb = (size_t)t * NP_ + (row0 + m * 16 + rb + r);
                    float xt   = X[xb * NDS_ + d * NS_];   // x[p, d, 0]
                    float mm   = Mm[xb * ND_ + d];
                    float diff = xt - pv;
                    s0 += diff * diff * mm;
                    s1 += fabsf(diff) * mm;
                    s2 += fabsf(diff) / (xt + EPS_) * mm;
                    s3 += mm;
                }
        }
    }
#pragma unroll
    for (int off = 16; off > 0; off >>= 1) {
        s0 += __shfl_xor(s0, off);
        s1 += __shfl_xor(s1, off);
        s2 += __shfl_xor(s2, off);
        s3 += __shfl_xor(s3, off);
    }
    if (lane == 0) {
        atomicAdd(&accum[0], s0);
        atomicAdd(&accum[1], s1);
        atomicAdd(&accum[2], s2);
        atomicAdd(&accum[3], s3);
    }
}

// Gates = Acat @ Wcat^T + (b_ih + b_hh)   (2048 x 1024), f32 out
__global__ void k_gemm_gates(const bf16_t* Acat, const bf16_t* W, const float* bih,
                             const float* bhh, float* G) {
    int lane = threadIdx.x & 31, wave = threadIdx.x >> 5;
    int row0 = (blockIdx.x * 8 + wave) * (MT_ * 16);
    int col0 = blockIdx.y * 32;
    int cn = lane & 15, rb = (lane >> 4) * 8;
    float b0 = bih[col0 + cn]      + bhh[col0 + cn];
    float b1 = bih[col0 + 16 + cn] + bhh[col0 + 16 + cn];
    v8f acc[MT_][2];
    gemm_4x2(Acat, KCAT_, W, KCAT_, row0, col0, KCAT_, b0, b1, acc, lane);
#pragma unroll
    for (int m = 0; m < MT_; ++m)
#pragma unroll
        for (int r = 0; r < 8; ++r) {
            size_t ro = (size_t)(row0 + m * 16 + rb + r) * G4H_ + col0 + cn;
            G[ro]      = acc[m][0][r];
            G[ro + 16] = acc[m][1][r];
        }
}

// Elementwise LSTM cell + scatter back into h/c at batch_idx rows (unique per step)
__global__ void k_lstm(const float* G, const int* bidx, int t, float* h, float* c, bf16_t* hb) {
    int p = blockIdx.x;
    int j = threadIdx.x;
    int b = bidx[t * NP_ + p];
    const float* g = G + (size_t)p * G4H_;
    float i_ = g[j], f_ = g[NH_ + j], gg = g[2 * NH_ + j], o_ = g[3 * NH_ + j];
    size_t o = (size_t)b * NH_ + j;
    float cn = sigm(f_) * c[o] + sigm(i_) * tanhf(gg);
    float hn = sigm(o_) * tanhf(cn);
    c[o]  = cn;
    h[o]  = hn;
    hb[o] = to_bf16(hn);
}

__global__ void k_final(const float* accum, float* out) {
    if (threadIdx.x == 0) {
        float tot = accum[3];
        out[0] = accum[0] / tot;
        out[1] = accum[1] / tot;
        out[2] = accum[2] / tot;
    }
}

// ---------------------------------------------------------------------------
// Host launcher
// ---------------------------------------------------------------------------
extern "C" void kernel_launch(void* const* d_in, const int* in_sizes, int n_in,
                              void* d_out, int out_size, void* d_ws, size_t ws_size,
                              hipStream_t stream) {
    (void)in_sizes; (void)n_in; (void)out_size; (void)ws_size;

    const float* X    = (const float*)d_in[0];
    const float* Mm   = (const float*)d_in[1];
    const int*   bidx = (const int*)  d_in[2];
    const float* Wih  = (const float*)d_in[3];
    const float* bih  = (const float*)d_in[4];
    const float* Whh  = (const float*)d_in[5];
    const float* bhh  = (const float*)d_in[6];
    const float* Wo1  = (const float*)d_in[7];
    const float* bo1  = (const float*)d_in[8];
    const float* Wo2  = (const float*)d_in[9];
    const float* bo2  = (const float*)d_in[10];
    const float* Wp1  = (const float*)d_in[11];
    const float* bp1  = (const float*)d_in[12];
    const float* Wp2  = (const float*)d_in[13];
    const float* bp2  = (const float*)d_in[14];
    float* out = (float*)d_out;

    // Workspace bump allocator (256B aligned). Total ~24 MB — L2-resident.
    char* wp = (char*)d_ws;
    auto alloc = [&](size_t bytes) -> void* {
        void* p = (void*)wp;
        wp += (bytes + 255) & ~(size_t)255;
        return p;
    };
    float*  h     = (float*) alloc((size_t)NB_ * NH_ * 4);
    float*  c     = (float*) alloc((size_t)NB_ * NH_ * 4);
    bf16_t* hb    = (bf16_t*)alloc((size_t)NB_ * NH_ * 2);
    bf16_t* ub    = (bf16_t*)alloc((size_t)NB_ * NH_ * 2);
    bf16_t* Acat  = (bf16_t*)alloc((size_t)NP_ * KCAT_ * 2);
    bf16_t* qb    = (bf16_t*)alloc((size_t)NP_ * NH_ * 2);
    float*  G     = (float*) alloc((size_t)NP_ * G4H_ * 4);
    bf16_t* Wo1b  = (bf16_t*)alloc((size_t)NH_ * NH_ * 2);
    bf16_t* Wo2b  = (bf16_t*)alloc((size_t)NH_ * NH_ * 2);
    bf16_t* Wp1b  = (bf16_t*)alloc((size_t)NH_ * NH_ * 2);
    bf16_t* Wp2b  = (bf16_t*)alloc((size_t)32 * NH_ * 2);
    bf16_t* Wcatb = (bf16_t*)alloc((size_t)G4H_ * KCAT_ * 2);
    float*  accum = (float*) alloc(4 * 4);

    // State init (h = c = 0, accumulators = 0) — deterministic each call.
    {
        int n = NB_ * NH_;
        k_init<<<(n + 255) / 256, 256, 0, stream>>>(h, c, hb, accum, n);
    }
    // Weight packing to bf16.
    {
        int n = G4H_ * KCAT_;  // largest range
        k_pack<<<(n + 255) / 256, 256, 0, stream>>>(Wo1, Wo2, Wp1, Wp2, Wih, Whh,
                                                    Wo1b, Wo2b, Wp1b, Wp2b, Wcatb);
    }

    for (int t = 0; t < NT_; ++t) {
        // ODE: u = tanh(h Wo1^T + bo1); h += DT*tanh(u Wo2^T + bo2)
        k_gemm_tanh<<<dim3(NB_ / ROWS_PER_BLOCK_, NH_ / 32), 256, 0, stream>>>(hb, Wo1b, bo1, ub);
        k_gemm_ode <<<dim3(NB_ / ROWS_PER_BLOCK_, NH_ / 32), 256, 0, stream>>>(ub, Wo2b, bo2, h, hb);
        // Gather [x_t | h[idx]] to bf16
        {
            int n = NP_ * KCAT_;
            k_gather<<<(n + 255) / 256, 256, 0, stream>>>(X, bidx, hb, Acat, t);
        }
        // Prediction + fused masked error reduction
        k_gemm_relu<<<dim3(NP_ / ROWS_PER_BLOCK_, NH_ / 32), 256, 0, stream>>>(Acat, Wp1b, bp1, qb);
        k_gemm_err <<<dim3(NP_ / ROWS_PER_BLOCK_, 1),        256, 0, stream>>>(qb, Wp2b, bp2, X, Mm, t, accum);
        // LSTM gates + cell update + scatter
        k_gemm_gates<<<dim3(NP_ / ROWS_PER_BLOCK_, G4H_ / 32), 256, 0, stream>>>(Acat, Wcatb, bih, bhh, G);
        k_lstm<<<NP_, NH_, 0, stream>>>(G, bidx, t, h, c, hb);
    }

    k_final<<<1, 32, 0, stream>>>(accum, out);
}
